// MFE_970662609464
// MI455X (gfx1250) — compile-verified, matrix operand order checked by reading, not run on hardware
//
#include <hip/hip_runtime.h>

// ---------------- constants ----------------
static constexpr int Bc = 2, Fc = 8, Cc = 3, IMGc = 224, Pc = 16;
static constexpr int DIMc = 512, HEADSc = 8, DHc = 64, DEPTHc = 4;
static constexpr int HPc = IMGc / Pc, WPc = IMGc / Pc;       // 14, 14
static constexpr int Nc_ = HPc * WPc;                        // 196
static constexpr int PDc = Cc * Pc * Pc;                     // 768
static constexpr int FFc = 4 * DIMc;                         // 2048
static constexpr int Tc = Bc * Fc * Nc_;                     // 3136 tokens (49*64)
static constexpr int QKVc = 3 * HEADSc * DHc;                // 1536
static constexpr int NPAD = 256;                             // 196 padded to tile multiple
static constexpr int Gs = Bc * HEADSc * Fc;                  // 128 spatial groups
static constexpr float SCALEc = 0.125f;                      // DH^-0.5
static constexpr float NEGF = -3.402823466e38f;
static constexpr float PIf = 3.14159265358979323846f;

typedef __attribute__((ext_vector_type(16))) _Float16 v16h;
typedef __attribute__((ext_vector_type(8)))  _Float16 v8h;
typedef __attribute__((ext_vector_type(8)))  float    v8f;

union FragH { v16h v; v8h h[2]; };

static __device__ inline _Float16 f2h(float x) { return (_Float16)x; }

// ---------------- generic batched GEMM: D[g] = A[g](MxK) * Bt[g](NcolxK)^T ----------------
// Requirements (guaranteed by caller): M % 64 == 0, Ncol % 256 == 0, K % 32 == 0.
// Block: 256 threads = 8 waves (2x4). Wave tile 32x64 (8 accumulators).
// Per ISA f16 fragment layout: lanes 0-15 hold row M=lane, K=[0..7],[16..23];
// lanes 16-31 hold row M=lane-16, K=[8..15],[24..31].
__global__ __launch_bounds__(256) void gemm_f16_wmma(
    const _Float16* __restrict__ A, const _Float16* __restrict__ Bt,
    float* __restrict__ D, _Float16* __restrict__ D16,
    const float* __restrict__ bias, const float* __restrict__ resid,
    int M, int Ncol, int K,
    long long sA, long long sB, long long sD, int gelu_flag) {
  const int g = blockIdx.z;
  const _Float16* Ag = A + (long long)g * sA;
  const _Float16* Bg = Bt + (long long)g * sB;
  const long long doff = (long long)g * sD;

  const int lane = threadIdx.x & 31;
  const int wave = threadIdx.x >> 5;
  const int wm = wave >> 2, wn = wave & 3;          // 2 x 4 wave grid
  const int m0 = blockIdx.y * 64 + wm * 32;
  const int n0 = blockIdx.x * 256 + wn * 64;
  const int koff = (lane >> 4) * 8;

  const _Float16* pa0 = Ag + (size_t)(m0 + (lane & 15)) * K + koff;
  const _Float16* pa1 = pa0 + (size_t)16 * K;
  const _Float16* pb0 = Bg + (size_t)(n0 + (lane & 15)) * K + koff;
  const _Float16* pb1 = pb0 + (size_t)16 * K;
  const _Float16* pb2 = pb0 + (size_t)32 * K;
  const _Float16* pb3 = pb0 + (size_t)48 * K;

  // warm the cache path once per stream (global_prefetch_b8)
  __builtin_prefetch(pa0, 0, 1);
  __builtin_prefetch(pa1, 0, 1);
  __builtin_prefetch(pb0, 0, 1);
  __builtin_prefetch(pb1, 0, 1);
  __builtin_prefetch(pb2, 0, 1);
  __builtin_prefetch(pb3, 0, 1);

  v8f acc[2][4] = {};

  for (int k0 = 0; k0 < K; k0 += 32) {
    FragH a0, a1, b0, b1, b2, b3;
    a0.h[0] = *(const v8h*)(pa0 + k0);  a0.h[1] = *(const v8h*)(pa0 + k0 + 16);
    a1.h[0] = *(const v8h*)(pa1 + k0);  a1.h[1] = *(const v8h*)(pa1 + k0 + 16);
    b0.h[0] = *(const v8h*)(pb0 + k0);  b0.h[1] = *(const v8h*)(pb0 + k0 + 16);
    b1.h[0] = *(const v8h*)(pb1 + k0);  b1.h[1] = *(const v8h*)(pb1 + k0 + 16);
    b2.h[0] = *(const v8h*)(pb2 + k0);  b2.h[1] = *(const v8h*)(pb2 + k0 + 16);
    b3.h[0] = *(const v8h*)(pb3 + k0);  b3.h[1] = *(const v8h*)(pb3 + k0 + 16);
    acc[0][0] = __builtin_amdgcn_wmma_f32_16x16x32_f16(false, a0.v, false, b0.v, (short)0, acc[0][0], false, false);
    acc[0][1] = __builtin_amdgcn_wmma_f32_16x16x32_f16(false, a0.v, false, b1.v, (short)0, acc[0][1], false, false);
    acc[0][2] = __builtin_amdgcn_wmma_f32_16x16x32_f16(false, a0.v, false, b2.v, (short)0, acc[0][2], false, false);
    acc[0][3] = __builtin_amdgcn_wmma_f32_16x16x32_f16(false, a0.v, false, b3.v, (short)0, acc[0][3], false, false);
    acc[1][0] = __builtin_amdgcn_wmma_f32_16x16x32_f16(false, a1.v, false, b0.v, (short)0, acc[1][0], false, false);
    acc[1][1] = __builtin_amdgcn_wmma_f32_16x16x32_f16(false, a1.v, false, b1.v, (short)0, acc[1][1], false, false);
    acc[1][2] = __builtin_amdgcn_wmma_f32_16x16x32_f16(false, a1.v, false, b2.v, (short)0, acc[1][2], false, false);
    acc[1][3] = __builtin_amdgcn_wmma_f32_16x16x32_f16(false, a1.v, false, b3.v, (short)0, acc[1][3], false, false);
  }

  const int nc = lane & 15;
  const int rshift = (lane >> 4) * 8;  // C/D layout: VGPR r -> M = r + 8*(lane>=16)
#pragma unroll
  for (int ti = 0; ti < 2; ++ti) {
#pragma unroll
    for (int tj = 0; tj < 4; ++tj) {
      int col = n0 + tj * 16 + nc;
      float bb = bias ? bias[col] : 0.f;
#pragma unroll
      for (int r = 0; r < 8; ++r) {
        int row = m0 + ti * 16 + rshift + r;
        long long idx = doff + (long long)row * Ncol + col;
        float v = acc[ti][tj][r] + bb;
        if (resid) v += resid[idx];
        if (gelu_flag) v = 0.5f * v * (1.f + erff(v * 0.70710678118654752f));
        if (D)   D[idx] = v;
        if (D16) D16[idx] = f2h(v);
      }
    }
  }
}

// ---------------- elementwise / prep kernels ----------------
__global__ void patchify_kernel(const float* __restrict__ seq, _Float16* __restrict__ Xp) {
  long long i = (long long)blockIdx.x * blockDim.x + threadIdx.x;
  if (i >= (long long)Tc * PDc) return;
  int col = (int)(i % PDc);
  long long row = i / PDc;
  int n = (int)(row % Nc_);
  long long bf = row / Nc_;            // b*F + f
  int hp = n / WPc, wp = n % WPc;
  int ph = col / (Pc * Cc);
  int r2 = col % (Pc * Cc);
  int pw = r2 / Cc, c = r2 % Cc;
  long long sidx = ((bf * Cc + c) * (long long)IMGc + (hp * Pc + ph)) * IMGc + (wp * Pc + pw);
  Xp[i] = f2h(seq[sidx]);
}

__global__ void keep_kernel(const float* __restrict__ obs, float* __restrict__ keep) {
  int i = blockIdx.x * blockDim.x + threadIdx.x;
  if (i >= Tc) return;
  int n = i % Nc_;
  int bf = i / Nc_;
  int hp = n / WPc, wp = n % WPc;
  float s = 0.f;
  for (int ph = 0; ph < Pc; ++ph)
    for (int pw = 0; pw < Pc; ++pw)
      s += obs[((long long)bf * IMGc + (hp * Pc + ph)) * IMGc + (wp * Pc + pw)];
  keep[i] = (s * (1.f / 256.f) > 0.5f) ? 1.f : 0.f;
}

__global__ void tcvt_kernel(const float* __restrict__ W, _Float16* __restrict__ Wt, int K, int Ncol) {
  long long i = (long long)blockIdx.x * blockDim.x + threadIdx.x;
  if (i >= (long long)K * Ncol) return;
  int nn = (int)(i / K);
  int k = (int)(i % K);
  Wt[i] = f2h(W[(long long)k * Ncol + nn]);
}

__global__ void cvt_kernel(const float* __restrict__ x, _Float16* __restrict__ y, long long n) {
  long long i = (long long)blockIdx.x * blockDim.x + threadIdx.x;
  if (i < n) y[i] = f2h(x[i]);
}

// wave32 layernorm: one wave per row of 512
__global__ __launch_bounds__(256) void ln_kernel(const float* __restrict__ x,
                                                 const float* __restrict__ w,
                                                 const float* __restrict__ b,
                                                 _Float16* __restrict__ out, int rows) {
  int wave = threadIdx.x >> 5, lane = threadIdx.x & 31;
  int row = blockIdx.x * 8 + wave;
  if (row >= rows) return;
  const float* xr = x + (size_t)row * DIMc;
  float vals[16];
  float sum = 0.f, sq = 0.f;
#pragma unroll
  for (int i = 0; i < 16; ++i) {
    float v = xr[lane + i * 32];
    vals[i] = v; sum += v; sq += v * v;
  }
#pragma unroll
  for (int m = 16; m > 0; m >>= 1) {
    sum += __shfl_xor(sum, m, 32);
    sq  += __shfl_xor(sq,  m, 32);
  }
  float mean = sum * (1.f / DIMc);
  float var = sq * (1.f / DIMc) - mean * mean;
  float inv = rsqrtf(var + 1e-5f);
  _Float16* orow = out + (size_t)row * DIMc;
#pragma unroll
  for (int i = 0; i < 16; ++i) {
    int c = lane + i * 32;
    orow[c] = f2h((vals[i] - mean) * inv * w[c] + b[c]);
  }
}

// fused temporal attention over F=8 per (b, head, n) group
__global__ __launch_bounds__(64) void attn_temporal_kernel(const float* __restrict__ qkv,
                                                           const float* __restrict__ keep,
                                                           _Float16* __restrict__ o) {
  int gid = blockIdx.x;                     // (b*HEADS + h)*N + n
  int n = gid % Nc_;
  int t = gid / Nc_;
  int h = t % HEADSc;
  int b = t / HEADSc;
  __shared__ float sq_[Fc][DHc], sk_[Fc][DHc], sv_[Fc][DHc], sp_[Fc][Fc];
  int tid = threadIdx.x;                    // 64 threads: d index
  for (int f = 0; f < Fc; ++f) {
    size_t rowb = (((size_t)(b * Fc + f)) * Nc_ + n) * QKVc;
    sq_[f][tid] = qkv[rowb + h * DHc + tid] * SCALEc;
    sk_[f][tid] = qkv[rowb + HEADSc * DHc + h * DHc + tid];
    sv_[f][tid] = qkv[rowb + 2 * HEADSc * DHc + h * DHc + tid];
  }
  __syncthreads();
  // rotary on (f, pair)
  for (int p = tid; p < Fc * 32; p += 64) {
    int f = p >> 5, pi = p & 31, d0 = 2 * pi;
    float ang = (float)f * expf(-((float)d0 / 64.f) * 9.210340371976184f); // 10000^{-d0/64}
    float s = sinf(ang), c = cosf(ang);
    float q0 = sq_[f][d0], q1 = sq_[f][d0 + 1];
    sq_[f][d0]     = q0 * c - q1 * s;
    sq_[f][d0 + 1] = q1 * c + q0 * s;
    float k0 = sk_[f][d0], k1 = sk_[f][d0 + 1];
    sk_[f][d0]     = k0 * c - k1 * s;
    sk_[f][d0 + 1] = k1 * c + k0 * s;
  }
  __syncthreads();
  {
    int i = tid >> 3, j = tid & 7;
    float s = 0.f;
    for (int d = 0; d < DHc; ++d) s += sq_[i][d] * sk_[j][d];
    if (j == i) s = NEGF;
    if (keep[((size_t)b * Fc + j) * Nc_ + n] == 0.f) s = NEGF;
    sp_[i][j] = s;
  }
  __syncthreads();
  if (tid < Fc) {
    float mx = NEGF;
    for (int j = 0; j < Fc; ++j) mx = fmaxf(mx, sp_[tid][j]);
    float sum = 0.f;
    for (int j = 0; j < Fc; ++j) { float e = expf(sp_[tid][j] - mx); sp_[tid][j] = e; sum += e; }
    float inv = 1.f / sum;
    for (int j = 0; j < Fc; ++j) sp_[tid][j] *= inv;
  }
  __syncthreads();
  for (int f = 0; f < Fc; ++f) {
    float acc = 0.f;
    for (int j = 0; j < Fc; ++j) acc += sp_[f][j] * sv_[j][tid];
    size_t tok = (size_t)(b * Fc + f) * Nc_ + n;
    o[tok * (HEADSc * DHc) + h * DHc + tid] = f2h(acc);
  }
}

// spatial prep: rotary+scale into padded Qs/Ks [G][NPAD][64]; V transposed+padded Vt [G][NPAD][NPAD]
// (only Vt rows d<64 and cols n<196 are non-zero)
__global__ __launch_bounds__(256) void spatial_prep_kernel(const float* __restrict__ qkv,
                                                           _Float16* __restrict__ Qs,
                                                           _Float16* __restrict__ Ks,
                                                           _Float16* __restrict__ Vt) {
  int n = blockIdx.x;                 // 0..NPAD-1
  int g = blockIdx.y;                 // (b*HEADS + h)*F + f
  int d = threadIdx.x;                // 0..255
  size_t voff = ((size_t)g * NPAD + d) * NPAD + n;
  if (d >= DHc) { Vt[voff] = (_Float16)0.f; return; }
  size_t qoff = ((size_t)g * NPAD + n) * DHc + d;
  if (n >= Nc_) {
    Qs[qoff] = (_Float16)0.f; Ks[qoff] = (_Float16)0.f; Vt[voff] = (_Float16)0.f;
    return;
  }
  int f = g % Fc;
  int t = g / Fc;
  int h = t % HEADSc;
  int b = t / HEADSc;
  size_t rowb = (((size_t)(b * Fc + f)) * Nc_ + n) * QKVc;
  int half = d >> 1;
  int hp = n / WPc, wp = n % WPc;
  float ang;
  if (half < 16) {
    float sc = 1.f + 4.f * (float)half / 15.f;
    ang = (-1.f + 2.f * (float)hp / 13.f) * sc * PIf;
  } else {
    float sc = 1.f + 4.f * (float)(half - 16) / 15.f;
    ang = (-1.f + 2.f * (float)wp / 13.f) * sc * PIf;
  }
  float s = sinf(ang), c = cosf(ang);
  int d0 = d & ~1;
  float q0 = qkv[rowb + h * DHc + d0],       q1 = qkv[rowb + h * DHc + d0 + 1];
  float k0 = qkv[rowb + 512 + h * DHc + d0], k1 = qkv[rowb + 512 + h * DHc + d0 + 1];
  float qr = (d & 1) ? (q1 * c + q0 * s) : (q0 * c - q1 * s);
  float kr = (d & 1) ? (k1 * c + k0 * s) : (k0 * c - k1 * s);
  Qs[qoff] = f2h(qr * SCALEc);
  Ks[qoff] = f2h(kr);
  Vt[voff] = f2h(qkv[rowb + 1024 + h * DHc + d]);
}

// spatial mask + softmax on f16 sim: one block per (g, row i)
__global__ __launch_bounds__(256) void spatial_softmax_kernel(const _Float16* __restrict__ sim,
                                                              const float* __restrict__ keep,
                                                              _Float16* __restrict__ P) {
  int i = blockIdx.x;          // row 0..NPAD-1
  int g = blockIdx.y;          // group
  int j = threadIdx.x;         // 0..255 == NPAD
  int f = g % Fc;
  int b = g / (Fc * HEADSc);
  size_t base = ((size_t)g * NPAD + i) * NPAD;
  float v = (float)sim[base + j];
  if (j >= Nc_ || j == i) v = NEGF;
  else if (keep[((size_t)b * Fc + f) * Nc_ + j] == 0.f) v = NEGF;
  __shared__ float red[256];
  red[threadIdx.x] = v;
  __syncthreads();
  for (int s = 128; s > 0; s >>= 1) {
    if (threadIdx.x < s) red[threadIdx.x] = fmaxf(red[threadIdx.x], red[threadIdx.x + s]);
    __syncthreads();
  }
  float mx = red[0];
  __syncthreads();
  float e = expf(v - mx);
  red[threadIdx.x] = e;
  __syncthreads();
  for (int s = 128; s > 0; s >>= 1) {
    if (threadIdx.x < s) red[threadIdx.x] += red[threadIdx.x + s];
    __syncthreads();
  }
  float inv = 1.f / red[0];
  P[base + j] = f2h(e * inv);
}

// gather spatial attention output back into token-major f16 matrix
__global__ void spatial_gather_kernel(const _Float16* __restrict__ Os, _Float16* __restrict__ o) {
  long long i = (long long)blockIdx.x * blockDim.x + threadIdx.x;
  if (i >= (long long)Tc * (HEADSc * DHc)) return;
  int col = (int)(i % (HEADSc * DHc));
  long long tok = i / (HEADSc * DHc);
  int h = col / DHc, d = col % DHc;
  int n = (int)(tok % Nc_);
  long long bf = tok / Nc_;
  int f = (int)(bf % Fc);
  int b = (int)(bf / Fc);
  int g = (b * HEADSc + h) * Fc + f;
  o[i] = Os[((size_t)g * NPAD + n) * NPAD + d];
}

__global__ void unpatchify_kernel(const float* __restrict__ U, float* __restrict__ out) {
  long long i = (long long)blockIdx.x * blockDim.x + threadIdx.x;
  if (i >= (long long)Bc * Fc * Cc * IMGc * IMGc) return;
  int x = (int)(i % IMGc);
  long long t1 = i / IMGc;
  int y = (int)(t1 % IMGc);
  long long t2 = t1 / IMGc;
  int c = (int)(t2 % Cc);
  long long t3 = t2 / Cc;
  int f = (int)(t3 % Fc);
  int b = (int)(t3 / Fc);
  int hp = y / Pc, ph = y % Pc, wp = x / Pc, pw = x % Pc;
  long long row = ((long long)(b * Fc + f)) * Nc_ + hp * WPc + wp;
  int col = ph * (Pc * Cc) + pw * Cc + c;
  out[i] = U[row * PDc + col];
}

// ---------------- host orchestration ----------------
extern "C" void kernel_launch(void* const* d_in, const int* in_sizes, int n_in,
                              void* d_out, int out_size, void* d_ws, size_t ws_size,
                              hipStream_t stream) {
  (void)in_sizes; (void)n_in; (void)out_size; (void)ws_size;
  const float* seq       = (const float*)d_in[0];
  const float* obs       = (const float*)d_in[1];
  const float* patch_w   = (const float*)d_in[2];
  const float* patch_b   = (const float*)d_in[3];
  const float* unpatch_w = (const float*)d_in[4];
  const float* unpatch_b = (const float*)d_in[5];
  const float* ln_t_w    = (const float*)d_in[6];
  const float* ln_t_b    = (const float*)d_in[7];
  const float* ln_s_w    = (const float*)d_in[8];
  const float* ln_s_b    = (const float*)d_in[9];
  const float* ln_f_w    = (const float*)d_in[10];
  const float* ln_f_b    = (const float*)d_in[11];
  const float* t_qkv     = (const float*)d_in[12];
  const float* t_out_w   = (const float*)d_in[13];
  const float* t_out_b   = (const float*)d_in[14];
  const float* s_qkv     = (const float*)d_in[15];
  const float* s_out_w   = (const float*)d_in[16];
  const float* s_out_b   = (const float*)d_in[17];
  const float* ff_w1     = (const float*)d_in[18];
  const float* ff_b1     = (const float*)d_in[19];
  const float* ff_w2     = (const float*)d_in[20];
  const float* ff_b2     = (const float*)d_in[21];

  char* base = (char*)d_ws;
  size_t off = 0;
  auto alloc = [&](size_t bytes) -> void* {
    off = (off + 255) & ~(size_t)255;
    void* p = base + off;
    off += bytes;
    return p;
  };

  _Float16* XpH   = (_Float16*)alloc((size_t)Tc * PDc * 2);
  float*    xF    = (float*)   alloc((size_t)Tc * DIMc * 4);
  _Float16* hH    = (_Float16*)alloc((size_t)Tc * DIMc * 2);
  float*    qkvF  = (float*)   alloc((size_t)Tc * QKVc * 4);
  _Float16* oH    = (_Float16*)alloc((size_t)Tc * DIMc * 2);
  _Float16* hffH  = (_Float16*)alloc((size_t)Tc * FFc * 2);
  float*    keepF = (float*)   alloc((size_t)Tc * 4);
  _Float16* Qs    = (_Float16*)alloc((size_t)Gs * NPAD * DHc * 2);
  _Float16* Ks    = (_Float16*)alloc((size_t)Gs * NPAD * DHc * 2);
  _Float16* Vt    = (_Float16*)alloc((size_t)Gs * NPAD * NPAD * 2);
  _Float16* simH  = (_Float16*)alloc((size_t)Gs * NPAD * NPAD * 2);
  _Float16* PH    = (_Float16*)alloc((size_t)Gs * NPAD * NPAD * 2);
  _Float16* OsH   = (_Float16*)alloc((size_t)Gs * NPAD * NPAD * 2);
  float*    UF    = (float*)   alloc((size_t)Tc * PDc * 4);
  _Float16* patchT   = (_Float16*)alloc((size_t)DIMc * PDc * 2);
  _Float16* unpatchT = (_Float16*)alloc((size_t)PDc * DIMc * 2);
  _Float16* qkvT     = (_Float16*)alloc((size_t)QKVc * DIMc * 2);
  _Float16* outT     = (_Float16*)alloc((size_t)DIMc * DIMc * 2);
  _Float16* ff1T     = (_Float16*)alloc((size_t)FFc * DIMc * 2);
  _Float16* ff2T     = (_Float16*)alloc((size_t)DIMc * FFc * 2);

  auto gemm = [&](const _Float16* A, const _Float16* Bt, float* D, _Float16* D16,
                  const float* bias, const float* resid, int M, int Ncol, int K,
                  long long sA, long long sB, long long sD, int G, int gelu) {
    dim3 grid(Ncol / 256, M / 64, G);
    gemm_f16_wmma<<<grid, dim3(256), 0, stream>>>(A, Bt, D, D16, bias, resid,
                                                  M, Ncol, K, sA, sB, sD, gelu);
  };
  auto tcvt = [&](const float* W, _Float16* Wt, int K, int Ncol) {
    long long tot = (long long)K * Ncol;
    tcvt_kernel<<<(unsigned)((tot + 255) / 256), 256, 0, stream>>>(W, Wt, K, Ncol);
  };
  auto ln = [&](const float* x, const float* w, const float* b, _Float16* out) {
    ln_kernel<<<(Tc + 7) / 8, 256, 0, stream>>>(x, w, b, out, Tc);
  };

  // ---- patch embed ----
  patchify_kernel<<<(unsigned)(((long long)Tc * PDc + 255) / 256), 256, 0, stream>>>(seq, XpH);
  keep_kernel<<<(Tc + 255) / 256, 256, 0, stream>>>(obs, keepF);
  tcvt(patch_w, patchT, PDc, DIMc);
  gemm(XpH, patchT, xF, nullptr, patch_b, nullptr, Tc, DIMc, PDc, 0, 0, 0, 1, 0);

  for (int l = 0; l < DEPTHc; ++l) {
    // ---- temporal attention ----
    ln(xF, ln_t_w + l * DIMc, ln_t_b + l * DIMc, hH);
    tcvt(t_qkv + (size_t)l * DIMc * QKVc, qkvT, DIMc, QKVc);
    gemm(hH, qkvT, qkvF, nullptr, nullptr, nullptr, Tc, QKVc, DIMc, 0, 0, 0, 1, 0);
    attn_temporal_kernel<<<Bc * HEADSc * Nc_, 64, 0, stream>>>(qkvF, keepF, oH);
    tcvt(t_out_w + (size_t)l * DIMc * DIMc, outT, DIMc, DIMc);
    gemm(oH, outT, xF, nullptr, t_out_b + l * DIMc, xF, Tc, DIMc, DIMc, 0, 0, 0, 1, 0);

    // ---- spatial attention ----
    ln(xF, ln_s_w + l * DIMc, ln_s_b + l * DIMc, hH);
    tcvt(s_qkv + (size_t)l * DIMc * QKVc, qkvT, DIMc, QKVc);
    gemm(hH, qkvT, qkvF, nullptr, nullptr, nullptr, Tc, QKVc, DIMc, 0, 0, 0, 1, 0);
    spatial_prep_kernel<<<dim3(NPAD, Gs), 256, 0, stream>>>(qkvF, Qs, Ks, Vt);
    // sim = Q K^T  (f16 out)
    gemm(Qs, Ks, nullptr, simH, nullptr, nullptr, NPAD, NPAD, DHc,
         (long long)NPAD * DHc, (long long)NPAD * DHc, (long long)NPAD * NPAD, Gs, 0);
    spatial_softmax_kernel<<<dim3(NPAD, Gs), 256, 0, stream>>>(simH, keepF, PH);
    // O = P V (f16 out; Vt zero-padded so cols >= 64 are zero)
    gemm(PH, Vt, nullptr, OsH, nullptr, nullptr, NPAD, NPAD, NPAD,
         (long long)NPAD * NPAD, (long long)NPAD * NPAD, (long long)NPAD * NPAD, Gs, 0);
    spatial_gather_kernel<<<(unsigned)(((long long)Tc * DIMc + 255) / 256), 256, 0, stream>>>(OsH, oH);
    tcvt(s_out_w + (size_t)l * DIMc * DIMc, outT, DIMc, DIMc);
    gemm(oH, outT, xF, nullptr, s_out_b + l * DIMc, xF, Tc, DIMc, DIMc, 0, 0, 0, 1, 0);

    // ---- feed-forward ----
    ln(xF, ln_f_w + l * DIMc, ln_f_b + l * DIMc, hH);
    tcvt(ff_w1 + (size_t)l * DIMc * FFc, ff1T, DIMc, FFc);
    gemm(hH, ff1T, nullptr, hffH, ff_b1 + l * FFc, nullptr, Tc, FFc, DIMc, 0, 0, 0, 1, 1);
    tcvt(ff_w2 + (size_t)l * FFc * DIMc, ff2T, FFc, DIMc);
    gemm(hffH, ff2T, xF, nullptr, ff_b2 + l * DIMc, xF, Tc, DIMc, FFc, 0, 0, 0, 1, 0);
  }

  // ---- unpatch ----
  cvt_kernel<<<(unsigned)(((long long)Tc * DIMc + 255) / 256), 256, 0, stream>>>(xF, hH, (long long)Tc * DIMc);
  tcvt(unpatch_w, unpatchT, DIMc, PDc);
  gemm(hH, unpatchT, UF, nullptr, unpatch_b, nullptr, Tc, PDc, DIMc, 0, 0, 0, 1, 0);
  unpatchify_kernel<<<(unsigned)(((long long)Bc * Fc * Cc * IMGc * IMGc + 255) / 256), 256, 0, stream>>>(
      UF, (float*)d_out);
}
